// Attention_48876727828685
// MI455X (gfx1250) — matrix operand order repaired and checked by reading, not compiled, not run on hardware
//
#include <hip/hip_runtime.h>
#include <math.h>

#define BATCH 8
#define SEQ   1024
#define CH    768
#define NHEADS 12
#define HD    64
#define C3    (3*CH)

#define HB1 8192          // top-13-bit histogram bins (32 KB LDS)
#define HB2 (1<<19)       // low-19-bit histogram bins (2 MB global)

typedef __attribute__((ext_vector_type(16))) __bf16 v16bf;
typedef __attribute__((ext_vector_type(8)))  float  v8f;

union FragAB { unsigned u[8]; v16bf v; };
union FragC  { float f[8]; v8f v; };

// fp32 -> bf16 round-to-nearest-even
static __device__ __forceinline__ unsigned short f2bf(float f) {
  unsigned u = __float_as_uint(f);
  u += 0x7FFFu + ((u >> 16) & 1u);
  return (unsigned short)(u >> 16);
}

static __device__ __forceinline__ void cvt_st4(unsigned short* d, float4 f,
                                               float thr, int useThr) {
  if (useThr) {
    f.x = (f.x >= thr) ? f.x : 0.0f;
    f.y = (f.y >= thr) ? f.y : 0.0f;
    f.z = (f.z >= thr) ? f.z : 0.0f;
    f.w = (f.w >= thr) ? f.w : 0.0f;
  }
  d[0] = f2bf(f.x); d[1] = f2bf(f.y); d[2] = f2bf(f.z); d[3] = f2bf(f.w);
}

// ---------------------------------------------------------------------------
// Generic bf16 WMMA GEMM: C[z] = alpha * (A[z] * B[z]) + cconst (+bias)(+resid)
//   A: fp32 [M][K] row-major (converted to bf16 in LDS staging; optional
//      threshold applied to A values for the attention*V pass)
//   B: fp32; bMode==0 -> stored [K][N] row-major (transposed while staging)
//            bMode==1 -> stored [N][K] row-major (direct staging)
//   Macro tile 128(M) x 64(N), K-step 32, 8 waves, each wave a 32x32 sub-tile
//   (2x2 fragments of v_wmma_f32_16x16x32_bf16).
// All dims here are multiples of the tile sizes (checked on host side).
// ---------------------------------------------------------------------------
__global__ __launch_bounds__(256)
void gemm_bf16_wmma(const float* __restrict__ A, long lda, long aZ,
                    const float* __restrict__ Bm, long ldb, long bZ,
                    float* __restrict__ Cm, long ldc, long cZ,
                    int K, int bMode, float alpha, float cconst,
                    const float* __restrict__ bias,
                    const float* __restrict__ resid, long residZ,
                    const float* __restrict__ thrPtr)
{
  __shared__ unsigned short As[128 * 32];   // 8 KB
  __shared__ unsigned short Bs[64 * 32];    // 4 KB

  const int z = blockIdx.z;
  A  += (long)z * aZ;
  Bm += (long)z * bZ;
  Cm += (long)z * cZ;
  if (resid) resid += (long)z * residZ;

  const int m0 = blockIdx.y * 128;
  const int n0 = blockIdx.x * 64;
  const int t    = threadIdx.x;
  const int lane = t & 31;
  const int wave = t >> 5;
  const int wm = (wave & 3) * 32;   // 4 row-groups  -> 128
  const int wn = (wave >> 2) * 32;  // 2 col-groups  -> 64

  const int useThr = (thrPtr != nullptr);
  float thr = 0.0f;
  if (useThr) thr = *thrPtr;

  FragC acc[2][2];
  v8f zero8 = {0.f,0.f,0.f,0.f,0.f,0.f,0.f,0.f};
#pragma unroll
  for (int i = 0; i < 2; ++i)
#pragma unroll
    for (int j = 0; j < 2; ++j) acc[i][j].v = zero8;

  // staging coordinates (fixed per thread)
  const int arow = t >> 1,  aseg = (t & 1) * 16;        // A: 128 x 32
  const int bkk  = t >> 3,  bcs  = (t & 7) * 8;         // B mode0: 32 x 64
  const int bcol = t >> 2,  bseg = (t & 3) * 8;         // B mode1: 64 x 32

  const int ksteps = K >> 5;
  for (int kt = 0; kt < ksteps; ++kt) {
    const int k0 = kt << 5;

    // ---- stage A (fp32 -> bf16, optional threshold) ----
    {
      const float* src = A + (long)(m0 + arow) * lda + k0 + aseg;
      float4 f0 = ((const float4*)src)[0];
      float4 f1 = ((const float4*)src)[1];
      float4 f2 = ((const float4*)src)[2];
      float4 f3 = ((const float4*)src)[3];
      unsigned short* d = &As[arow * 32 + aseg];
      cvt_st4(d + 0,  f0, thr, useThr);
      cvt_st4(d + 4,  f1, thr, useThr);
      cvt_st4(d + 8,  f2, thr, useThr);
      cvt_st4(d + 12, f3, thr, useThr);
      if (kt + 1 < ksteps)              // prefetch next A K-tile
        __builtin_prefetch(src + 32, 0, 0);
    }
    // ---- stage B into Bs[col][k] (bf16) ----
    if (bMode == 0) {   // B stored [K][N]: transpose while staging
      const float* src = Bm + (long)(k0 + bkk) * ldb + n0 + bcs;
      float4 f0 = ((const float4*)src)[0];
      float4 f1 = ((const float4*)src)[1];
      float vals[8] = {f0.x,f0.y,f0.z,f0.w,f1.x,f1.y,f1.z,f1.w};
#pragma unroll
      for (int c = 0; c < 8; ++c) Bs[(bcs + c) * 32 + bkk] = f2bf(vals[c]);
      if (kt + 1 < ksteps)
        __builtin_prefetch(src + 32 * ldb, 0, 0);
    } else {            // B stored [N][K]: direct staging
      const float* src = Bm + (long)(n0 + bcol) * ldb + k0 + bseg;
      float4 f0 = ((const float4*)src)[0];
      float4 f1 = ((const float4*)src)[1];
      unsigned short* d = &Bs[bcol * 32 + bseg];
      cvt_st4(d + 0, f0, 0.f, 0);
      cvt_st4(d + 4, f1, 0.f, 0);
      if (kt + 1 < ksteps)
        __builtin_prefetch(src + 32, 0, 0);
    }
    __syncthreads();

    // ---- fragment loads (ISA 7.12.2 layouts) + WMMA ----
    const int hi = lane >> 4;
    const int rb = lane & 15;
    FragAB afr[2], bfr[2];
#pragma unroll
    for (int i = 0; i < 2; ++i) {
      const unsigned short* ap = &As[(wm + i * 16 + rb) * 32];
#pragma unroll
      for (int j = 0; j < 8; ++j) {
        const int kb = (j < 4 ? 2 * j : 16 + 2 * (j - 4)) + hi * 8;
        afr[i].u[j] = *(const unsigned*)&ap[kb];
      }
    }
#pragma unroll
    for (int i = 0; i < 2; ++i) {
      const unsigned short* bp = &Bs[(wn + i * 16 + rb) * 32];
#pragma unroll
      for (int j = 0; j < 8; ++j) {
        const int kb = (j < 4 ? 2 * j : 16 + 2 * (j - 4)) + hi * 8;
        bfr[i].u[j] = *(const unsigned*)&bp[kb];
      }
    }
#pragma unroll
    for (int i = 0; i < 2; ++i)
#pragma unroll
      for (int j = 0; j < 2; ++j)
        acc[i][j].v = __builtin_amdgcn_wmma_f32_16x16x32_bf16(
            false, afr[i].v, false, bfr[j].v,
            (short)0, acc[i][j].v, false, false);
    __syncthreads();
  }

  // ---- epilogue (C layout: VGPR r -> M = r + 8*hi, lane&15 -> N) ----
  const int hi = lane >> 4;
#pragma unroll
  for (int i = 0; i < 2; ++i)
#pragma unroll
    for (int j = 0; j < 2; ++j) {
      const int gc = n0 + wn + j * 16 + (lane & 15);
      const float bv = bias ? bias[gc] : 0.0f;
#pragma unroll
      for (int r = 0; r < 8; ++r) {
        const int gr = m0 + wm + i * 16 + r + hi * 8;
        float v = acc[i][j].f[r] * alpha + cconst + bv;
        if (resid) v += resid[(long)gr * ldc + gc];
        Cm[(long)gr * ldc + gc] = v;
      }
    }
}

// ---------------------------------------------------------------------------
// Row softmax over attn[12288][1024] (in place) + LDS histogram of the
// top-13 float bits of each probability (all probs are positive -> IEEE bit
// order == numeric order). One wave per row; block handles 128 rows.
// ---------------------------------------------------------------------------
__global__ __launch_bounds__(256)
void softmax_hist_kernel(float* __restrict__ attn, unsigned* __restrict__ hist1)
{
  __shared__ unsigned hist[HB1];
  for (int i = threadIdx.x; i < HB1; i += 256) hist[i] = 0;
  __syncthreads();

  const int wave = threadIdx.x >> 5;
  const int lane = threadIdx.x & 31;

  for (int it = 0; it < 16; ++it) {
    const int row = blockIdx.x * 128 + it * 8 + wave;
    float* p = attn + (long)row * SEQ;
    float4 vals[8];
    float mx = -3.4e38f;
#pragma unroll
    for (int i = 0; i < 8; ++i) {
      vals[i] = ((const float4*)p)[lane + i * 32];
      mx = fmaxf(mx, fmaxf(fmaxf(vals[i].x, vals[i].y),
                           fmaxf(vals[i].z, vals[i].w)));
    }
#pragma unroll
    for (int s = 16; s > 0; s >>= 1) mx = fmaxf(mx, __shfl_xor(mx, s, 32));

    float sum = 0.0f;
#pragma unroll
    for (int i = 0; i < 8; ++i) {
      vals[i].x = __expf(vals[i].x - mx);
      vals[i].y = __expf(vals[i].y - mx);
      vals[i].z = __expf(vals[i].z - mx);
      vals[i].w = __expf(vals[i].w - mx);
      sum += vals[i].x + vals[i].y + vals[i].z + vals[i].w;
    }
#pragma unroll
    for (int s = 16; s > 0; s >>= 1) sum += __shfl_xor(sum, s, 32);
    const float inv = 1.0f / sum;

#pragma unroll
    for (int i = 0; i < 8; ++i) {
      vals[i].x *= inv; vals[i].y *= inv; vals[i].z *= inv; vals[i].w *= inv;
      ((float4*)p)[lane + i * 32] = vals[i];
      atomicAdd(&hist[__float_as_uint(vals[i].x) >> 19], 1u);
      atomicAdd(&hist[__float_as_uint(vals[i].y) >> 19], 1u);
      atomicAdd(&hist[__float_as_uint(vals[i].z) >> 19], 1u);
      atomicAdd(&hist[__float_as_uint(vals[i].w) >> 19], 1u);
    }
  }
  __syncthreads();
  for (int i = threadIdx.x; i < HB1; i += 256) {
    const unsigned c = hist[i];
    if (c) atomicAdd(&hist1[i], c);
  }
}

// Find the 13-bit bin containing the Krank-th largest value (descending).
__global__ void select_bin1_kernel(const unsigned* __restrict__ hist1,
                                   unsigned* __restrict__ sel, unsigned Krank)
{
  __shared__ unsigned csum[256];
  const int t = threadIdx.x;
  unsigned s = 0;
  for (int i = 0; i < HB1 / 256; ++i) s += hist1[t * (HB1 / 256) + i];
  csum[t] = s;
  __syncthreads();
  if (t == 0) {
    unsigned cum = 0; int chunk = 0;
    for (int c = 255; c >= 0; --c) {
      if (cum + csum[c] >= Krank) { chunk = c; break; }
      cum += csum[c];
    }
    unsigned rem = Krank - cum, bin = 0;
    for (int b2 = chunk * (HB1 / 256) + (HB1 / 256) - 1;
         b2 >= chunk * (HB1 / 256); --b2) {
      const unsigned h = hist1[b2];
      if (h >= rem) { bin = (unsigned)b2; break; }
      rem -= h;
    }
    sel[0] = bin;   // top-13 bits
    sel[1] = rem;   // remaining rank within the bin (1-based)
  }
}

// Histogram of the low 19 bits for probabilities whose top-13 bits == sel[0].
__global__ __launch_bounds__(256)
void hist2_kernel(const float* __restrict__ attn,
                  const unsigned* __restrict__ sel,
                  unsigned* __restrict__ hist2)
{
  const unsigned bin1 = sel[0];
  const long base = (long)blockIdx.x * 2048;
#pragma unroll
  for (int i = 0; i < 8; ++i) {
    const unsigned u = __float_as_uint(attn[base + i * 256 + threadIdx.x]);
    if ((u >> 19) == bin1) atomicAdd(&hist2[u & (HB2 - 1)], 1u);
  }
}

// Resolve the exact 32-bit threshold value.
__global__ void select_thr_kernel(const unsigned* __restrict__ hist2,
                                  const unsigned* __restrict__ sel,
                                  float* __restrict__ thrOut)
{
  __shared__ unsigned csum[256];
  const int t = threadIdx.x;
  unsigned s = 0;
  for (int i = 0; i < HB2 / 256; ++i) s += hist2[t * (HB2 / 256) + i];
  csum[t] = s;
  __syncthreads();
  if (t == 0) {
    const unsigned K2 = sel[1];
    unsigned cum = 0; int chunk = 0;
    for (int c = 255; c >= 0; --c) {
      if (cum + csum[c] >= K2) { chunk = c; break; }
      cum += csum[c];
    }
    unsigned rem = K2 - cum, low = 0;
    for (int b2 = chunk * (HB2 / 256) + (HB2 / 256) - 1;
         b2 >= chunk * (HB2 / 256); --b2) {
      const unsigned h = hist2[b2];
      if (h >= rem) { low = (unsigned)b2; break; }
      rem -= h;
    }
    *thrOut = __uint_as_float((sel[0] << 19) | low);
  }
}

__global__ void zero_u32_kernel(unsigned* __restrict__ p, long n)
{
  const long i = (long)blockIdx.x * blockDim.x + threadIdx.x;
  if (i < n) p[i] = 0;
}

// ---------------------------------------------------------------------------
extern "C" void kernel_launch(void* const* d_in, const int* in_sizes, int n_in,
                              void* d_out, int out_size, void* d_ws, size_t ws_size,
                              hipStream_t stream)
{
  const float* x      = (const float*)d_in[0];   // [8,1024,768]
  const float* origin = (const float*)d_in[1];   // [8,1024,768]
  const float* W_qkv  = (const float*)d_in[2];   // [768,2304]
  const float* W_proj = (const float*)d_in[3];   // [768,768]
  const float* b_proj = (const float*)d_in[4];   // [768]
  float* out = (float*)d_out;

  // workspace layout
  float*    qkv   = (float*)d_ws;                              // 8*1024*2304 f32
  float*    attn  = qkv  + (size_t)BATCH * SEQ * C3;           // 12*1024*1024 f32
  float*    oh    = attn + (size_t)NHEADS * SEQ * SEQ;         // 1024*768 f32
  unsigned* hist1 = (unsigned*)(oh + (size_t)SEQ * CH);        // 8192
  unsigned* hist2 = hist1 + HB1;                               // 524288
  unsigned* sel   = hist2 + HB2;                               // 2
  float*    thrp  = (float*)(sel + 2);                         // 1

  const float scale = 0.125f;                               // 64^-0.5
  const float cbias = logf(1.0f + (float)SEQ / 197.0f);
  const unsigned total = (unsigned)NHEADS * SEQ * SEQ;
  const unsigned Krank = (unsigned)((double)total * 0.2);   // kidx + 1

  const dim3 blk(256);

  // 1) QKV projection: [8192 x 768] * [768 x 2304]
  gemm_bf16_wmma<<<dim3(C3 / 64, (BATCH * SEQ) / 128, 1), blk, 0, stream>>>(
      x, CH, 0, W_qkv, C3, 0, qkv, C3, 0, CH, /*bMode=*/0,
      1.0f, 0.0f, nullptr, nullptr, 0, nullptr);

  const long zhn = (long)(HB1 + HB2);
  for (int b = 0; b < BATCH; ++b) {
    const float* qb = qkv + (size_t)b * SEQ * C3;

    zero_u32_kernel<<<dim3((unsigned)((zhn + 255) / 256)), blk, 0, stream>>>(hist1, zhn);

    // 2) logits[h] = scale * q[h] k[h]^T + cbias   (B given as [N][K] rows)
    gemm_bf16_wmma<<<dim3(SEQ / 64, SEQ / 128, NHEADS), blk, 0, stream>>>(
        qb, C3, HD, qb + CH, C3, HD, attn, SEQ, (long)SEQ * SEQ, HD, /*bMode=*/1,
        scale, cbias, nullptr, nullptr, 0, nullptr);

    // 3) softmax rows + histogram
    softmax_hist_kernel<<<dim3((NHEADS * SEQ) / 128), blk, 0, stream>>>(attn, hist1);

    // 4) exact rank selection (two-level radix select on positive floats)
    select_bin1_kernel<<<1, 256, 0, stream>>>(hist1, sel, Krank);
    hist2_kernel<<<dim3(total / 2048), blk, 0, stream>>>(attn, sel, hist2);
    select_thr_kernel<<<1, 256, 0, stream>>>(hist2, sel, thrp);

    // 5) out_heads[h] = threshold(attn[h]) @ v[h]   (threshold in A staging)
    gemm_bf16_wmma<<<dim3(HD / 64, SEQ / 128, NHEADS), blk, 0, stream>>>(
        attn, SEQ, (long)SEQ * SEQ, qb + 2 * CH, C3, HD, oh, CH, HD, SEQ, /*bMode=*/0,
        1.0f, 0.0f, nullptr, nullptr, 0, thrp);

    // 6) out = oh @ W_proj + b_proj + origin
    gemm_bf16_wmma<<<dim3(CH / 64, SEQ / 128, 1), blk, 0, stream>>>(
        oh, CH, 0, W_proj, CH, 0, out + (size_t)b * SEQ * CH, CH, 0, CH, /*bMode=*/0,
        1.0f, 0.0f, b_proj, origin + (size_t)b * SEQ * CH, 0, nullptr);
  }
}
